// MoE_68324339745450
// MI455X (gfx1250) — matrix operand order repaired and checked by reading, not compiled, not run on hardware
//
#include <hip/hip_runtime.h>

typedef __bf16 bf16;
typedef __attribute__((ext_vector_type(4)))  bf16  v4bf;
typedef __attribute__((ext_vector_type(8)))  bf16  v8bf;
typedef __attribute__((ext_vector_type(16))) bf16  v16bf;
typedef __attribute__((ext_vector_type(8)))  float v8f;

#define T_DIM   2048
#define T_INTER 1024
#define T_NEXP  8
#define T_TOPK  2
#define T_TOK   4096
#define TILE    64
#define KSTEP   64
#define LDST    72   // padded LDS row stride in bf16 (144B: 16B-aligned, conflict-free)
#define TILEB   (TILE * LDST)   // elements per LDS tile buffer

// ---------------- workspace layout (bytes) ----------------
static const size_t OFF_XB   = 0;                         // T*DIM bf16        = 16,777,216
static const size_t OFF_W1B  = OFF_XB   + 16777216;       // E*INTER*DIM bf16  = 33,554,432
static const size_t OFF_W3B  = OFF_W1B  + 33554432;
static const size_t OFF_W2B  = OFF_W3B  + 33554432;       // E*DIM*INTER bf16
static const size_t OFF_WS1B = OFF_W2B  + 33554432;       // INTER*DIM bf16    = 4,194,304
static const size_t OFF_WS3B = OFF_WS1B + 4194304;
static const size_t OFF_WS2B = OFF_WS3B + 4194304;
static const size_t OFF_G    = OFF_WS2B + 4194304;        // (T*K)*INTER bf16  = 16,777,216
static const size_t OFF_GS   = OFF_G    + 16777216;       // T*INTER bf16      = 8,388,608
static const size_t OFF_O    = OFF_GS   + 8388608;        // (T*K)*DIM f32     = 67,108,864
static const size_t OFF_WBUF = OFF_O    + 67108864;       // T*K f32
static const size_t OFF_IDX  = OFF_WBUF + 32768;          // T*K int
static const size_t OFF_BKT  = OFF_IDX  + 32768;          // E*T int
static const size_t OFF_CUR  = OFF_BKT  + 131072;         // E int (cursor == counts)

// ---------------- CDNA5 async copy: global -> LDS (ASYNCcnt) ----------------
__device__ __forceinline__ void async_ld16(unsigned int lds_off, const void* gptr) {
  asm volatile("global_load_async_to_lds_b128 %0, %1, off"
               :: "v"(lds_off), "v"(gptr) : "memory");
}
__device__ __forceinline__ void wait_async0() {
  asm volatile("s_wait_asynccnt 0x0" ::: "memory");
}
__device__ __forceinline__ unsigned int lds_addr(const void* p) {
  return (unsigned int)(size_t)p;   // generic LDS pointer: low 32 bits = wave-relative LDS offset
}

// ---------------- fp32 -> bf16 conversion ----------------
__global__ __launch_bounds__(256) void cvt_kernel(const float* __restrict__ s,
                                                  bf16* __restrict__ d, int n4) {
  int i = blockIdx.x * blockDim.x + threadIdx.x;
  if (i < n4) {
    float4 v = ((const float4*)s)[i];
    v4bf o;
    o[0] = (bf16)v.x; o[1] = (bf16)v.y; o[2] = (bf16)v.z; o[3] = (bf16)v.w;
    ((v4bf*)d)[i] = o;
  }
}

// ---------------- gate: logits -> softmax -> top-2 ----------------
__global__ __launch_bounds__(256) void gate_topk(const float* __restrict__ x,
                                                 const float* __restrict__ gw,
                                                 int* __restrict__ idxbuf,
                                                 float* __restrict__ wbuf) {
  const int lane = threadIdx.x & 31;
  const int t = blockIdx.x * 8 + (threadIdx.x >> 5);
  const float* xr = x + (size_t)t * T_DIM;
  float p[T_NEXP];
#pragma unroll
  for (int e = 0; e < T_NEXP; ++e) p[e] = 0.f;
  for (int d = lane; d < T_DIM; d += 32) {
    float xv = xr[d];
#pragma unroll
    for (int e = 0; e < T_NEXP; ++e) p[e] += xv * gw[e * T_DIM + d];
  }
#pragma unroll
  for (int e = 0; e < T_NEXP; ++e) {
#pragma unroll
    for (int off = 16; off > 0; off >>= 1) p[e] += __shfl_xor(p[e], off, 32);
  }
  if (lane == 0) {
    float mx = p[0];
#pragma unroll
    for (int e = 1; e < T_NEXP; ++e) mx = fmaxf(mx, p[e]);
    float s = 0.f;
#pragma unroll
    for (int e = 0; e < T_NEXP; ++e) { p[e] = __expf(p[e] - mx); s += p[e]; }
    float inv = 1.0f / s;
    int e0 = 0; float b0 = p[0];
#pragma unroll
    for (int e = 1; e < T_NEXP; ++e) if (p[e] > b0) { b0 = p[e]; e0 = e; }
    int e1 = 0; float b1v = -1.f;
#pragma unroll
    for (int e = 0; e < T_NEXP; ++e) if (e != e0 && p[e] > b1v) { b1v = p[e]; e1 = e; }
    idxbuf[t * 2]     = e0; wbuf[t * 2]     = b0  * inv;
    idxbuf[t * 2 + 1] = e1; wbuf[t * 2 + 1] = b1v * inv;
  }
}

__global__ void zero_cursor(int* __restrict__ cursor) {
  if (threadIdx.x < T_NEXP) cursor[threadIdx.x] = 0;
}

// slot s = t*2+k -> bucket[expert]. Order in bucket is nondeterministic but the
// per-slot math and output locations are order-invariant -> deterministic output.
__global__ __launch_bounds__(256) void scatter_slots(const int* __restrict__ idxbuf,
                                                     int* __restrict__ cursor,
                                                     int* __restrict__ bucket) {
  int s = blockIdx.x * blockDim.x + threadIdx.x;
  if (s < T_TOK * T_TOPK) {
    int e = idxbuf[s];
    int pos = atomicAdd(&cursor[e], 1);
    bucket[e * T_TOK + pos] = s;
  }
}

// ---------------- WMMA fragment loads from LDS ----------------
// A (16x32 bf16): lane<16 -> row=lane&15, K {0..7,16..23}; lane>=16 -> K {8..15,24..31}
__device__ __forceinline__ v16bf frag_a(const bf16* s, int row, int lane, int kk) {
  int c0 = kk + (lane >> 4) * 8;
  v8bf lo = *(const v8bf*)(s + row * LDST + c0);
  v8bf hi = *(const v8bf*)(s + row * LDST + 16 + c0);
  v16bf a;
#pragma unroll
  for (int i = 0; i < 8; ++i) { a[i] = lo[i]; a[i + 8] = hi[i]; }
  return a;
}
// B (32x16 bf16): col=lane&15; lane<16 -> K {0..7,16..23}; lane>=16 -> K {8..15,24..31}
__device__ __forceinline__ v16bf frag_b(const bf16* s, int col, int lane, int kk) {
  int c0 = kk + (lane >> 4) * 8;
  v8bf lo = *(const v8bf*)(s + col * LDST + c0);
  v8bf hi = *(const v8bf*)(s + col * LDST + c0 + 16);
  v16bf b;
#pragma unroll
  for (int i = 0; i < 8; ++i) { b[i] = lo[i]; b[i + 8] = hi[i]; }
  return b;
}

// ---------------- up-projection: G = silu(X W1^T) * (X W3^T), gathered rows ----------------
__global__ __launch_bounds__(256) void expert_up(
    const bf16* __restrict__ xb, const bf16* __restrict__ w1b, const bf16* __restrict__ w3b,
    bf16* __restrict__ Gout, const int* __restrict__ bucket, const int* __restrict__ counts,
    int dense) {
  __shared__ bf16 Xs[2][TILEB], W1s[2][TILEB], W3s[2][TILEB];
  __shared__ int Ss[TILE];
  const int e = blockIdx.z;
  const int count = dense ? T_TOK : counts[e];
  const int row0 = blockIdx.y * TILE;
  if (row0 >= count) return;
  const int tid = threadIdx.x;
  if (tid < TILE) {
    int r = row0 + tid; if (r > count - 1) r = count - 1;
    Ss[tid] = dense ? r : bucket[e * T_TOK + r];
  }
  __syncthreads();
  const int lane = tid & 31, wid = tid >> 5;
  const int waveM = wid >> 2, waveN = wid & 3;
  const int gcol0 = blockIdx.x * TILE;
  const bf16* w1p = w1b + ((size_t)e * T_INTER + gcol0) * T_DIM;
  const bf16* w3p = w3b + ((size_t)e * T_INTER + gcol0) * T_DIM;
  const int lr = tid >> 2;          // staging row 0..63
  const int lc = (tid & 3) * 8;     // staging chunk 0/8/16/24 (+32 for second half)

  int tok = Ss[lr]; if (!dense) tok >>= 1;   // slot -> token
  const bf16* xsrc  = xb  + (size_t)tok * T_DIM + lc;
  const bf16* w1src = w1p + (size_t)lr  * T_DIM + lc;
  const bf16* w3src = w3p + (size_t)lr  * T_DIM + lc;
  unsigned int xds[2], w1ds[2], w3ds[2];
#pragma unroll
  for (int b = 0; b < 2; ++b) {
    xds[b]  = lds_addr(&Xs[b][lr * LDST + lc]);
    w1ds[b] = lds_addr(&W1s[b][lr * LDST + lc]);
    w3ds[b] = lds_addr(&W3s[b][lr * LDST + lc]);
  }

  v8f zero = {0.f, 0.f, 0.f, 0.f, 0.f, 0.f, 0.f, 0.f};
  v8f acc1[2] = {zero, zero}, acc3[2] = {zero, zero};

  // prologue: stage 0 into buffer 0 (two 32-elem chunks per 64-wide tile row segment)
  async_ld16(xds[0],      xsrc);       async_ld16(xds[0]  + 64, xsrc  + 32);
  async_ld16(w1ds[0],     w1src);      async_ld16(w1ds[0] + 64, w1src + 32);
  async_ld16(w3ds[0],     w3src);      async_ld16(w3ds[0] + 64, w3src + 32);

  const int NK = T_DIM / KSTEP;
#pragma clang loop unroll(disable)
  for (int i = 0; i < NK; ++i) {
    wait_async0();        // this wave's DMA for stage i has landed
    __syncthreads();      // everyone's stage-i data in LDS; prev-stage frag reads drained
    if (i + 1 < NK) {     // prefetch stage i+1 into the buffer freed at stage i-1
      int k = (i + 1) * KSTEP;
      int b = (i + 1) & 1;
      async_ld16(xds[b],       xsrc  + k);  async_ld16(xds[b]  + 64, xsrc  + k + 32);
      async_ld16(w1ds[b],      w1src + k);  async_ld16(w1ds[b] + 64, w1src + k + 32);
      async_ld16(w3ds[b],      w3src + k);  async_ld16(w3ds[b] + 64, w3src + k + 32);
    }
    const bf16* Xb  = Xs[i & 1];
    const bf16* W1b = W1s[i & 1];
    const bf16* W3b = W3s[i & 1];
#pragma unroll
    for (int kk = 0; kk < KSTEP; kk += 32) {
      v16bf b1 = frag_b(W1b, waveN * 16 + (lane & 15), lane, kk);
      v16bf b3 = frag_b(W3b, waveN * 16 + (lane & 15), lane, kk);
#pragma unroll
      for (int m = 0; m < 2; ++m) {
        v16bf a = frag_a(Xb, waveM * 32 + m * 16 + (lane & 15), lane, kk);
        acc1[m] = __builtin_amdgcn_wmma_f32_16x16x32_bf16(false, a, false, b1, (short)0, acc1[m], false, false);
        acc3[m] = __builtin_amdgcn_wmma_f32_16x16x32_bf16(false, a, false, b3, (short)0, acc3[m], false, false);
      }
    }
  }
  const int nloc = lane & 15, lh = lane >> 4;
#pragma unroll
  for (int m = 0; m < 2; ++m) {
#pragma unroll
    for (int j = 0; j < 8; ++j) {
      int rloc = waveM * 32 + m * 16 + j + lh * 8;   // C: vgpr j, lanes0-15 M=j, lanes16-31 M=j+8
      if (row0 + rloc < count) {
        float h1 = acc1[m][j], h3 = acc3[m][j];
        float g = (h1 / (1.0f + __expf(-h1))) * h3;  // silu(h1)*h3
        int slot = Ss[rloc];
        Gout[(size_t)slot * T_INTER + gcol0 + waveN * 16 + nloc] = (bf16)g;
      }
    }
  }
}

// ---------------- down-projection: Out[slot] = w_slot * (G W2^T) ----------------
__global__ __launch_bounds__(256) void expert_down(
    const bf16* __restrict__ Gin, const bf16* __restrict__ w2b,
    float* __restrict__ Out, const int* __restrict__ bucket, const int* __restrict__ counts,
    const float* __restrict__ wbuf, int dense) {
  __shared__ bf16 As[2][TILEB], Bs[2][TILEB];
  __shared__ int Ss[TILE];
  const int e = blockIdx.z;
  const int count = dense ? T_TOK : counts[e];
  const int row0 = blockIdx.y * TILE;
  if (row0 >= count) return;
  const int tid = threadIdx.x;
  if (tid < TILE) {
    int r = row0 + tid; if (r > count - 1) r = count - 1;
    Ss[tid] = dense ? r : bucket[e * T_TOK + r];
  }
  __syncthreads();
  const int lane = tid & 31, wid = tid >> 5;
  const int waveM = wid >> 2, waveN = wid & 3;
  const int gcol0 = blockIdx.x * TILE;
  const bf16* w2p = w2b + ((size_t)e * T_DIM + gcol0) * T_INTER;
  const int lr = tid >> 2;
  const int lc = (tid & 3) * 8;

  int arow = Ss[lr];   // routed: G indexed by slot; dense: Gs indexed by token
  const bf16* asrc = Gin + (size_t)arow * T_INTER + lc;
  const bf16* bsrc = w2p + (size_t)lr   * T_INTER + lc;
  unsigned int ads[2], bds[2];
#pragma unroll
  for (int b = 0; b < 2; ++b) {
    ads[b] = lds_addr(&As[b][lr * LDST + lc]);
    bds[b] = lds_addr(&Bs[b][lr * LDST + lc]);
  }

  v8f zero = {0.f, 0.f, 0.f, 0.f, 0.f, 0.f, 0.f, 0.f};
  v8f acc[2] = {zero, zero};

  async_ld16(ads[0],      asrc);      async_ld16(ads[0] + 64, asrc + 32);
  async_ld16(bds[0],      bsrc);      async_ld16(bds[0] + 64, bsrc + 32);

  const int NK = T_INTER / KSTEP;
#pragma clang loop unroll(disable)
  for (int i = 0; i < NK; ++i) {
    wait_async0();
    __syncthreads();
    if (i + 1 < NK) {
      int k = (i + 1) * KSTEP;
      int b = (i + 1) & 1;
      async_ld16(ads[b],      asrc + k);  async_ld16(ads[b] + 64, asrc + k + 32);
      async_ld16(bds[b],      bsrc + k);  async_ld16(bds[b] + 64, bsrc + k + 32);
    }
    const bf16* Ab = As[i & 1];
    const bf16* Bb = Bs[i & 1];
#pragma unroll
    for (int kk = 0; kk < KSTEP; kk += 32) {
      v16bf b = frag_b(Bb, waveN * 16 + (lane & 15), lane, kk);
#pragma unroll
      for (int m = 0; m < 2; ++m) {
        v16bf a = frag_a(Ab, waveM * 32 + m * 16 + (lane & 15), lane, kk);
        acc[m] = __builtin_amdgcn_wmma_f32_16x16x32_bf16(false, a, false, b, (short)0, acc[m], false, false);
      }
    }
  }
  const int nloc = lane & 15, lh = lane >> 4;
#pragma unroll
  for (int m = 0; m < 2; ++m) {
#pragma unroll
    for (int j = 0; j < 8; ++j) {
      int rloc = waveM * 32 + m * 16 + j + lh * 8;
      if (row0 + rloc < count) {
        int slot = Ss[rloc];
        float v = acc[m][j];
        if (!dense) v *= wbuf[slot];
        Out[(size_t)slot * T_DIM + gcol0 + waveN * 16 + nloc] = v;
      }
    }
  }
}

// ---------------- final combine: out = z_shared + O[t,0] + O[t,1] ----------------
__global__ __launch_bounds__(256) void combine(float* __restrict__ out,
                                               const float* __restrict__ O) {
  int i = blockIdx.x * blockDim.x + threadIdx.x;   // over T*DIM
  int t = i / T_DIM, c = i - t * T_DIM;
  out[i] += O[(size_t)(2 * t) * T_DIM + c] + O[(size_t)(2 * t + 1) * T_DIM + c];
}

extern "C" void kernel_launch(void* const* d_in, const int* in_sizes, int n_in,
                              void* d_out, int out_size, void* d_ws, size_t ws_size,
                              hipStream_t stream) {
  const float* x   = (const float*)d_in[0];
  const float* gw  = (const float*)d_in[1];
  const float* w1  = (const float*)d_in[2];
  const float* w2  = (const float*)d_in[3];
  const float* w3  = (const float*)d_in[4];
  const float* ws1 = (const float*)d_in[5];
  const float* ws2 = (const float*)d_in[6];
  const float* ws3 = (const float*)d_in[7];
  float* out = (float*)d_out;
  char* ws = (char*)d_ws;

  bf16* xb   = (bf16*)(ws + OFF_XB);
  bf16* w1b  = (bf16*)(ws + OFF_W1B);
  bf16* w3b  = (bf16*)(ws + OFF_W3B);
  bf16* w2b  = (bf16*)(ws + OFF_W2B);
  bf16* ws1b = (bf16*)(ws + OFF_WS1B);
  bf16* ws3b = (bf16*)(ws + OFF_WS3B);
  bf16* ws2b = (bf16*)(ws + OFF_WS2B);
  bf16* G    = (bf16*)(ws + OFF_G);
  bf16* Gs   = (bf16*)(ws + OFF_GS);
  float* O     = (float*)(ws + OFF_O);
  float* wbuf  = (float*)(ws + OFF_WBUF);
  int* idxbuf  = (int*)(ws + OFF_IDX);
  int* bucket  = (int*)(ws + OFF_BKT);
  int* cursor  = (int*)(ws + OFF_CUR);   // cursor == per-expert counts after scatter

  auto cvt = [&](const float* s, bf16* d, int n) {
    int n4 = n / 4;
    cvt_kernel<<<(n4 + 255) / 256, 256, 0, stream>>>(s, d, n4);
  };
  cvt(x,   xb,   T_TOK * T_DIM);
  cvt(w1,  w1b,  T_NEXP * T_INTER * T_DIM);
  cvt(w2,  w2b,  T_NEXP * T_DIM * T_INTER);
  cvt(w3,  w3b,  T_NEXP * T_INTER * T_DIM);
  cvt(ws1, ws1b, T_INTER * T_DIM);
  cvt(ws2, ws2b, T_DIM * T_INTER);
  cvt(ws3, ws3b, T_INTER * T_DIM);

  gate_topk<<<T_TOK / 8, 256, 0, stream>>>(x, gw, idxbuf, wbuf);
  zero_cursor<<<1, 32, 0, stream>>>(cursor);
  scatter_slots<<<(T_TOK * T_TOPK) / 256, 256, 0, stream>>>(idxbuf, cursor, bucket);

  // routed experts
  expert_up<<<dim3(T_INTER / TILE, T_TOK / TILE, T_NEXP), 256, 0, stream>>>(
      xb, w1b, w3b, G, bucket, cursor, 0);
  // shared expert (dense over all tokens)
  expert_up<<<dim3(T_INTER / TILE, T_TOK / TILE, 1), 256, 0, stream>>>(
      xb, ws1b, ws3b, Gs, nullptr, nullptr, 1);

  expert_down<<<dim3(T_DIM / TILE, T_TOK / TILE, T_NEXP), 256, 0, stream>>>(
      G, w2b, O, bucket, cursor, wbuf, 0);
  expert_down<<<dim3(T_DIM / TILE, T_TOK / TILE, 1), 256, 0, stream>>>(
      Gs, ws2b, out, nullptr, nullptr, nullptr, 1);

  combine<<<(T_TOK * T_DIM) / 256, 256, 0, stream>>>(out, O);
}
